// MultiheadAttention_8461085573400
// MI455X (gfx1250) — matrix operand order repaired and checked by reading, not compiled, not run on hardware
//
#include <hip/hip_runtime.h>
#include <hip/hip_bf16.h>
#include <math.h>

// MI455X / gfx1250, wave32. bf16 WMMA pipeline:
//   f32 -> bf16 once; bf16 GEMMs (WMMA, f32 accum); V written transposed;
//   flash attention: QK^T WMMA, ds_swizzle max, row-sum via ones-WMMA, PV WMMA.

typedef __attribute__((ext_vector_type(16))) __bf16 v16bf;
typedef __attribute__((ext_vector_type(8)))  __bf16 v8bf;
typedef __attribute__((ext_vector_type(4)))  __bf16 v4bf;
typedef __attribute__((ext_vector_type(2)))  __bf16 v2bf;
typedef __attribute__((ext_vector_type(8)))  float  v8f;

#define B_  4
#define S_  2048
#define D_  1024
#define H_  16
#define DH_ 64

// v_exp_f32 (exp2) -- device builtin, avoids host glibc __exp2f collision.
static __device__ inline float fast_exp2(float x) {
  return __builtin_amdgcn_exp2f(x);
}

static __device__ inline v8f wmma_bf16(v16bf a, v16bf b, v8f c) {
  return __builtin_amdgcn_wmma_f32_16x16x32_bf16(
      false, a, false, b, (short)0, c, false, false);
}

static __device__ inline v16bf cat8(v8bf lo, v8bf hi) {
  return __builtin_shufflevector(lo, hi, 0, 1, 2, 3, 4, 5, 6, 7,
                                 8, 9, 10, 11, 12, 13, 14, 15);
}

// A-matrix 16x32 bf16 fragment: lane&15 = row M; e -> k = (e&7)+(e>=8?16:0)+8*half.
static __device__ inline v16bf load_a_bf16(const __bf16* __restrict__ rowp,
                                           int k0, int half) {
  const v8bf lo = *(const v8bf*)(rowp + k0 + half * 8);
  const v8bf hi = *(const v8bf*)(rowp + k0 + half * 8 + 16);
  return cat8(lo, hi);
}

// B-matrix 32x16 bf16 fragment: lane&15 = col N; e -> k = (e&7)+(e>=8?16:0)+8*half.
static __device__ inline v16bf load_b_bf16(const __bf16* __restrict__ colp,
                                           int k0, int half) {
  const v8bf lo = *(const v8bf*)(colp + k0 + half * 8);
  const v8bf hi = *(const v8bf*)(colp + k0 + half * 8 + 16);
  return cat8(lo, hi);
}

// XOR lane swizzle via DS_SWIZZLE group-of-32 mode: imm = (xor<<10) | 0x1F.
template <int IMM>
static __device__ inline float ds_swz_f32(float x) {
  return __int_as_float(__builtin_amdgcn_ds_swizzle(__float_as_int(x), IMM));
}
// Max over the 16-lane group (xor 1,2,4,8 stay within each half-wave group).
static __device__ inline float group16_max(float x) {
  x = fmaxf(x, ds_swz_f32<(1 << 10) | 0x1F>(x));
  x = fmaxf(x, ds_swz_f32<(2 << 10) | 0x1F>(x));
  x = fmaxf(x, ds_swz_f32<(4 << 10) | 0x1F>(x));
  x = fmaxf(x, ds_swz_f32<(8 << 10) | 0x1F>(x));
  return x;
}

// One-shot elementwise f32 -> bf16 (n multiple of 4).
__global__ void __launch_bounds__(256)
cvt_f32_to_bf16(const float* __restrict__ src, __bf16* __restrict__ dst, int n) {
  const int i = (blockIdx.x * blockDim.x + threadIdx.x) * 4;
  if (i >= n) return;
  const float4 v = *(const float4*)(src + i);
  v4bf o;
  o[0] = (__bf16)v.x; o[1] = (__bf16)v.y; o[2] = (__bf16)v.z; o[3] = (__bf16)v.w;
  *(v4bf*)(dst + i) = o;
}

// Y = X[M,K] @ W[N,K]^T + bias. Wave tile 32(M) x 64(N), 2x4 WMMA per K-step.
// MODE 0: f32 row-major. MODE 1: bf16 row-major. MODE 2: bf16 transposed
// ([B,H,DH,S] given row = b*S+s, col = h*DH+d  ->  index (b*D+col)*S + s).
template <int MODE>
__global__ void __launch_bounds__(128)
gemm_bf16(const __bf16* __restrict__ X, const __bf16* __restrict__ W,
          const float* __restrict__ bias, void* __restrict__ Yv,
          int M, int N, int K) {
  const int lane = threadIdx.x & 31;
  const int half = lane >> 4;
  const int n16  = lane & 15;
  const int wid  = blockIdx.x * (blockDim.x >> 5) + (threadIdx.x >> 5);
  const int tn_cnt = N >> 6;
  const int tm = wid / tn_cnt;
  const int tn = wid % tn_cnt;
  if (tm * 32 >= M) return;
  const int m0 = tm * 32, n0 = tn * 64;

  const __bf16* arow0 = X + (size_t)(m0 + n16) * K;
  const __bf16* arow1 = X + (size_t)(m0 + 16 + n16) * K;
  const __bf16* bcol[4];
#pragma unroll
  for (int j = 0; j < 4; ++j) bcol[j] = W + (size_t)(n0 + j * 16 + n16) * K;

  v8f c[2][4];
#pragma unroll
  for (int i = 0; i < 2; ++i)
#pragma unroll
    for (int j = 0; j < 4; ++j) { v8f z = {}; c[i][j] = z; }

  for (int k0 = 0; k0 < K; k0 += 32) {
    __builtin_prefetch(arow0 + k0 + 64, 0, 1);     // global_prefetch_b8
    __builtin_prefetch(bcol[0] + k0 + 64, 0, 1);
    const v16bf a0 = load_a_bf16(arow0, k0, half);
    const v16bf a1 = load_a_bf16(arow1, k0, half);
#pragma unroll
    for (int j = 0; j < 4; ++j) {
      const v16bf bj = load_b_bf16(bcol[j], k0, half);
      c[0][j] = wmma_bf16(a0, bj, c[0][j]);
      c[1][j] = wmma_bf16(a1, bj, c[1][j]);
    }
  }

  const int batch = m0 / S_;                       // MODE 2 only (tiles never straddle)
  const int s0    = m0 - batch * S_;
#pragma unroll
  for (int j = 0; j < 4; ++j) {
    const int col = n0 + j * 16 + n16;
    const float bn = bias[col];
#pragma unroll
    for (int i = 0; i < 2; ++i) {
#pragma unroll
      for (int r = 0; r < 8; ++r) {                // C layout: row=r+8*half, col=n16
        const float val = c[i][j][r] + bn;
        if (MODE == 0) {
          const size_t row = (size_t)(m0 + i * 16 + r + 8 * half);
          ((float*)Yv)[row * N + col] = val;
        } else if (MODE == 1) {
          const size_t row = (size_t)(m0 + i * 16 + r + 8 * half);
          ((__bf16*)Yv)[row * N + col] = (__bf16)val;
        } else {
          const size_t s = (size_t)(s0 + i * 16 + r + 8 * half);
          ((__bf16*)Yv)[(size_t)(batch * D_ + col) * S_ + s] = (__bf16)val;
        }
      }
    }
  }
}

// Flash attention. Q/K bf16 [B,S,D] row-major; V bf16 transposed [B*D, S];
// bf16 out [B,S,D]. One wave per (b,h,16-query tile); 32 keys per step.
// Within a window, key numbering is interleaved: phys p <-> key0 + p with
// s_lo col n = key0+2n, s_hi col n = key0+2n+1 (permutation-safe for softmax/PV).
__global__ void __launch_bounds__(128)
flash_attn_fwd(const __bf16* __restrict__ Q, const __bf16* __restrict__ Km,
               const __bf16* __restrict__ Vt, __bf16* __restrict__ O) {
  __shared__ __bf16 lds_p[4][16][40];              // 16x32 P per wave, 80B row stride
  const int lane  = threadIdx.x & 31;
  const int half  = lane >> 4;
  const int n16   = lane & 15;
  const int wslot = threadIdx.x >> 5;
  const int wid   = blockIdx.x * 4 + wslot;
  const int qt = wid & (S_ / 16 - 1);
  const int bh = wid / (S_ / 16);
  const int h  = bh & (H_ - 1);
  const int b  = bh / H_;
  const int q0 = qt * 16;

  const __bf16* qrow = Q + (size_t)(b * S_ + q0 + n16) * D_ + h * DH_;
  const v16bf aq0 = load_a_bf16(qrow, 0, half);
  const v16bf aq1 = load_a_bf16(qrow, 32, half);

  v16bf ones;
#pragma unroll
  for (int e = 0; e < 16; ++e) ones[e] = (__bf16)1.0f;

  float mrow[8], lrow[8], alpha[8];
#pragma unroll
  for (int r = 0; r < 8; ++r) { mrow[r] = -INFINITY; lrow[r] = 0.0f; }
  v8f o[4];
#pragma unroll
  for (int nt = 0; nt < 4; ++nt) { v8f z = {}; o[nt] = z; }

  const float SC = 0.125f * 1.44269504f;           // (1/sqrt(DH)) * log2(e)
  const int jp_last = qt >> 1;
  for (int jp = 0; jp <= jp_last; ++jp) {
    const int key0 = jp * 32;
    // QK^T: lane col n -> K rows key0+2n (lo) and key0+2n+1 (hi).
    const __bf16* krow_lo = Km + (size_t)(b * S_ + key0 + 2 * n16) * D_ + h * DH_;
    const __bf16* krow_hi = krow_lo + D_;
    v8f s_lo = {}, s_hi = {};
    s_lo = wmma_bf16(aq0, load_b_bf16(krow_lo, 0, half), s_lo);
    s_lo = wmma_bf16(aq1, load_b_bf16(krow_lo, 32, half), s_lo);
    s_hi = wmma_bf16(aq0, load_b_bf16(krow_hi, 0, half), s_hi);
    s_hi = wmma_bf16(aq1, load_b_bf16(krow_hi, 32, half), s_hi);

#pragma unroll
    for (int r = 0; r < 8; ++r) { s_lo[r] *= SC; s_hi[r] *= SC; }
    if (jp == jp_last) {                           // causal mask (wave-uniform branch)
#pragma unroll
      for (int r = 0; r < 8; ++r) {
        const int row = q0 + r + 8 * half;
        const int cl  = key0 + 2 * n16;
        if (cl > row)     s_lo[r] = -1.0e30f;
        if (cl + 1 > row) s_hi[r] = -1.0e30f;
      }
    }

    // Per-row max (ds_swizzle butterfly), exp2, packed P store to LDS.
#pragma unroll
    for (int r = 0; r < 8; ++r) {
      const float xl = s_lo[r], xh = s_hi[r];
      const float mx   = group16_max(fmaxf(xl, xh));
      const float mnew = fmaxf(mrow[r], mx);
      alpha[r] = fast_exp2(mrow[r] - mnew);
      mrow[r]  = mnew;
      v2bf pp;
      pp[0] = (__bf16)fast_exp2(xl - mnew);        // phys col 2*n16   (key0+2n)
      pp[1] = (__bf16)fast_exp2(xh - mnew);        // phys col 2*n16+1 (key0+2n+1)
      *(v2bf*)&lds_p[wslot][r + 8 * half][2 * n16] = pp;
    }

    // P as A-frag: two 16B LDS vector loads (phys k contiguous runs).
    const __bf16* prow = &lds_p[wslot][n16][0];
    const v16bf pa = cat8(*(const v8bf*)(prow + half * 16),
                          *(const v8bf*)(prow + half * 16 + 8));

    // Row sums via ones-WMMA: srow[r] = sum_k P[row][k], broadcast across group.
    v8f srow = {};
    srow = wmma_bf16(pa, ones, srow);
#pragma unroll
    for (int r = 0; r < 8; ++r) {
      lrow[r] = lrow[r] * alpha[r] + srow[r];
      o[0][r] *= alpha[r]; o[1][r] *= alpha[r];
      o[2][r] *= alpha[r]; o[3][r] *= alpha[r];
    }

    // O += P . V: B col n = dim d, element e -> phys key kp = e+16*half,
    // V^T rows contiguous in key -> two 16B loads per fragment.
#pragma unroll
    for (int nt = 0; nt < 4; ++nt) {
      const __bf16* vb = Vt + (size_t)(b * D_ + h * DH_ + nt * 16 + n16) * S_
                       + key0 + 16 * half;
      const v16bf bv = cat8(*(const v8bf*)(vb), *(const v8bf*)(vb + 8));
      o[nt] = wmma_bf16(pa, bv, o[nt]);
    }
  }

#pragma unroll
  for (int r = 0; r < 8; ++r) {
    const float inv = 1.0f / lrow[r];
    const size_t row = (size_t)(b * S_ + q0 + r + 8 * half);
#pragma unroll
    for (int nt = 0; nt < 4; ++nt)
      O[row * D_ + h * DH_ + nt * 16 + n16] = (__bf16)(o[nt][r] * inv);
  }
}

extern "C" void kernel_launch(void* const* d_in, const int* in_sizes, int n_in,
                              void* d_out, int out_size, void* d_ws, size_t ws_size,
                              hipStream_t stream) {
  (void)in_sizes; (void)n_in; (void)out_size; (void)ws_size;
  const float* x_q   = (const float*)d_in[0];
  const float* x_kv  = (const float*)d_in[1];
  const float* W_q   = (const float*)d_in[2];
  const float* b_q   = (const float*)d_in[3];
  const float* W_k   = (const float*)d_in[4];
  const float* b_k   = (const float*)d_in[5];
  const float* W_v   = (const float*)d_in[6];
  const float* b_v   = (const float*)d_in[7];
  const float* W_out = (const float*)d_in[8];
  const float* b_out = (const float*)d_in[9];
  float* out = (float*)d_out;

  const size_t act = (size_t)B_ * S_ * D_;
  const size_t wsz = (size_t)D_ * D_;
  __bf16* xqb  = (__bf16*)d_ws;
  __bf16* xkvb = xqb + act;
  __bf16* wqb  = xkvb + act;
  __bf16* wkb  = wqb + wsz;
  __bf16* wvb  = wkb + wsz;
  __bf16* wob  = wvb + wsz;
  __bf16* qb   = wob + wsz;
  __bf16* kb   = qb + act;
  __bf16* vtb  = kb + act;                         // transposed V [B*D, S]
  __bf16* ab   = vtb + act;

  {
    const int tpb = 256;
    const int ga = (int)(act / 4 / tpb);
    const int gw = (int)(wsz / 4 / tpb);
    cvt_f32_to_bf16<<<dim3(ga), dim3(tpb), 0, stream>>>(x_q,   xqb,  (int)act);
    cvt_f32_to_bf16<<<dim3(ga), dim3(tpb), 0, stream>>>(x_kv,  xkvb, (int)act);
    cvt_f32_to_bf16<<<dim3(gw), dim3(tpb), 0, stream>>>(W_q,   wqb,  (int)wsz);
    cvt_f32_to_bf16<<<dim3(gw), dim3(tpb), 0, stream>>>(W_k,   wkb,  (int)wsz);
    cvt_f32_to_bf16<<<dim3(gw), dim3(tpb), 0, stream>>>(W_v,   wvb,  (int)wsz);
    cvt_f32_to_bf16<<<dim3(gw), dim3(tpb), 0, stream>>>(W_out, wob,  (int)wsz);
  }

  const int M = B_ * S_;
  dim3 blk(128);
  const int gemm_waves = (M / 32) * (D_ / 64);     // 4096
  dim3 ggrid(gemm_waves / 4);

  gemm_bf16<1><<<ggrid, blk, 0, stream>>>(xqb,  wqb, b_q, qb,  M, D_, D_);
  gemm_bf16<1><<<ggrid, blk, 0, stream>>>(xkvb, wkb, b_k, kb,  M, D_, D_);
  gemm_bf16<2><<<ggrid, blk, 0, stream>>>(xkvb, wvb, b_v, vtb, M, D_, D_);

  const int attn_waves = B_ * H_ * (S_ / 16);      // 8192
  flash_attn_fwd<<<dim3(attn_waves / 4), blk, 0, stream>>>(qb, kb, vtb, ab);

  gemm_bf16<0><<<ggrid, blk, 0, stream>>>(ab, wob, b_out, out, M, D_, D_);
}
